// SingleAttention_90744069030107
// MI455X (gfx1250) — matrix operand order repaired and checked by reading, not compiled, not run on hardware
//
#include <hip/hip_runtime.h>

// ---------------------------------------------------------------------------
// Single-head attention for MI455X (gfx1250, wave32, WMMA bf16 16x16x32).
//   x:[4,2048,1024]f32, Wq/Wk/Wv:[1024,1024]f32  ->  out:[4,2048,1024]f32
// ~120 GFLOP total, all tensors L2-resident (192MB) -> compute bound ->
// everything through v_wmma bf16, with TDM (tensor_load_to_lds) streaming
// B-operand tiles into LDS double-buffers overlapped with WMMA.
// ---------------------------------------------------------------------------

typedef __attribute__((ext_vector_type(16))) __bf16 bf16x16;
typedef __attribute__((ext_vector_type(8)))  float  f32x8;
typedef __attribute__((ext_vector_type(4)))  unsigned int u32x4;
typedef __attribute__((ext_vector_type(8)))  int i32x8;
typedef __attribute__((ext_vector_type(4)))  int i32x4;

__device__ __forceinline__ unsigned short f2bf(float f) {
  unsigned u = __builtin_bit_cast(unsigned, f);
  u += 0x7FFFu + ((u >> 16) & 1u);               // round-to-nearest-even
  return (unsigned short)(u >> 16);
}
__device__ __forceinline__ float bf2f(unsigned short h) {
  unsigned u = ((unsigned)h) << 16;
  return __builtin_bit_cast(float, u);
}

__device__ __forceinline__ f32x8 wmma_bf16(bf16x16 a, bf16x16 b, f32x8 c) {
  return __builtin_amdgcn_wmma_f32_16x16x32_bf16(false, a, false, b,
                                                 (short)0, c, false, false);
}

// A fragment (16x32 bf16, row-major, leading dim `ld` elems):
// lane L: row = row0+(L&15); elems 0..7 -> k0+8h.., elems 8..15 -> k0+16+8h..
__device__ __forceinline__ bf16x16 load_frag_a(const unsigned short* base,
                                               int ld, int row0, int k0,
                                               int lane) {
  const int r = row0 + (lane & 15);
  const int h = lane >> 4;
  const unsigned short* p = base + (size_t)r * ld + k0 + 8 * h;
  union { bf16x16 v; uint4 q[2]; } u;
  u.q[0] = *(const uint4*)(p);
  u.q[1] = *(const uint4*)(p + 16);
  return u.v;
}

// B fragment (32x16): source stored k-contiguous per output column
// (row index = output column). lane L: col=col0+(L&15); elems 0..7 -> k0+8h.., elems 8..15 -> k0+16+8h..
__device__ __forceinline__ bf16x16 load_frag_b(const unsigned short* base,
                                               size_t ld, int col0, size_t k0,
                                               int lane) {
  const int c = col0 + (lane & 15);
  const int h = lane >> 4;
  const unsigned short* p = base + (size_t)c * ld + k0 + 8 * h;
  union { bf16x16 v; uint4 q[2]; } u;
  u.q[0] = *(const uint4*)(p);
  u.q[1] = *(const uint4*)(p + 16);
  return u.v;
}

// ---------------------------------------------------------------------------
// TDM: async 2-D tile load (global -> LDS), bf16 elements.
// D# per CDNA5 ISA ch.8: group0 {count=1, lds_addr, global_addr, type=2},
// group1 {data_size=2B, tensor_dim0=row_len, tile_dim0/1, dim0_stride}.
// Tracked on TENSORcnt. EXEC-independent; issued once per executing wave.
// 6-arg builtin form (clang-23 signature): (g0, g1, g2, g3, g4, cpol).
// ---------------------------------------------------------------------------
__device__ __forceinline__ void tdm_load_tile_2d(const void* lds_ptr,
                                                 const void* gptr,
                                                 unsigned tile_w,   // elems
                                                 unsigned tile_h,   // rows
                                                 unsigned long long stride) {
  unsigned lds_off =
      (unsigned)__builtin_amdgcn_readfirstlane((int)(size_t)lds_ptr);
  unsigned long long ga = (unsigned long long)(size_t)gptr;
  unsigned ga_lo = (unsigned)__builtin_amdgcn_readfirstlane((int)(unsigned)ga);
  unsigned ga_hi =
      (unsigned)__builtin_amdgcn_readfirstlane((int)(unsigned)(ga >> 32));

  u32x4 g0 = {1u,                                   // count=1 (valid user D#)
              lds_off,                              // lds_addr (bytes)
              ga_lo,                                // global_addr[31:0]
              (ga_hi & 0x1ffffffu) | 0x80000000u};  // ga[56:32] | type=2

  const unsigned td0 = (unsigned)stride;   // tensor_dim0 = full row length
  const unsigned td1 = 0x100000u;          // tensor_dim1: large (in-bounds)
  i32x8 g1 = {(int)0x00010000u,                         // data_size=1 (2B)
              (int)((td0 & 0xffffu) << 16),             // tensor_dim0 lo
              (int)(((td0 >> 16) & 0xffffu) |           // tensor_dim0 hi
                    ((td1 & 0xffffu) << 16)),           // tensor_dim1 lo
              (int)(((td1 >> 16) & 0xffffu) |           // tensor_dim1 hi
                    ((tile_w & 0xffffu) << 16)),        // tile_dim0
              (int)(tile_h & 0xffffu),                  // tile_dim1 (dim2=0)
              (int)(unsigned)(stride & 0xffffffffu),    // dim0_stride lo
              (int)(unsigned)((stride >> 32) & 0xffffu),// dim0_stride hi
              0};
  i32x4 gz4 = {0, 0, 0, 0};          // 2-D tile: groups 2/3 unused
  i32x8 gz8 = {0, 0, 0, 0, 0, 0, 0, 0};
  __builtin_amdgcn_tensor_load_to_lds(g0, g1, gz4, gz4, gz8, 0);
}

// ---------------------------------------------------------------------------
// Stage 0: casts
// ---------------------------------------------------------------------------
__global__ void sattn_cast_x(const float* __restrict__ x,
                             unsigned short* __restrict__ xb, int n) {
  int i = blockIdx.x * blockDim.x + threadIdx.x;
  if (i < n) xb[i] = f2bf(x[i]);
}

// W is [k][n] f32; produce Wt[n][k] bf16 (k-contiguous for B frags).
__global__ void sattn_cast_wt(const float* __restrict__ Wq,
                              const float* __restrict__ Wk,
                              const float* __restrict__ Wv,
                              unsigned short* __restrict__ Wtq,
                              unsigned short* __restrict__ Wtk,
                              unsigned short* __restrict__ Wtv) {
  const int sel = blockIdx.z;
  const float* W = sel == 0 ? Wq : (sel == 1 ? Wk : Wv);
  unsigned short* Wt = sel == 0 ? Wtq : (sel == 1 ? Wtk : Wtv);
  int i = blockIdx.x * blockDim.x + threadIdx.x;  // over 1024*1024
  int n = i >> 10, k = i & 1023;
  Wt[i] = f2bf(W[k * 1024 + n]);
}

// ---------------------------------------------------------------------------
// Stage 1: QKV GEMM.  C[8192x1024] = X[8192x1024] * W[1024x1024] (x3).
// Block = 8 waves sharing one 64-col N-strip; Wt strip TDM-staged into LDS
// in 64x256 bf16 chunks (32KB), double-buffered: B loaded from global ONCE
// per block (8x traffic reduction) and consumed via ds_load_b128.
// Each wave owns one 16-row M-tile; A-frags for a whole 256-deep chunk are
// preloaded (8 frags) so WMMAs don't stall on loadcnt.
// ---------------------------------------------------------------------------
__global__ __launch_bounds__(256) void sattn_qkv_gemm(
    const unsigned short* __restrict__ X,
    const unsigned short* __restrict__ Wtq,
    const unsigned short* __restrict__ Wtk,
    const unsigned short* __restrict__ Wtv,
    unsigned short* __restrict__ Q,
    unsigned short* __restrict__ K,
    unsigned short* __restrict__ Vt) {
  __shared__ __align__(16) unsigned short WtS[2][64 * 256];  // 2 x 32KB

  const int sel = blockIdx.z;
  const unsigned short* Wt = sel == 0 ? Wtq : (sel == 1 ? Wtk : Wtv);
  const int lane = threadIdx.x & 31;
  const int wave = threadIdx.x >> 5;
  const int nstrip = blockIdx.x & 15;          // 16 strips of 64 columns
  const int n0 = nstrip << 6;
  const int m0 = (((blockIdx.x >> 4) << 3) + wave) << 4;  // 512 row tiles
  const unsigned short* Wstrip = Wt + (size_t)n0 * 1024;

  f32x8 acc[4] = {};
  if (wave == 0) tdm_load_tile_2d(&WtS[0][0], Wstrip, 256, 64, 1024);

  for (int c = 0; c < 4; ++c) {            // four 256-deep K chunks
    if (wave == 0) {
      if (c < 3) {
        tdm_load_tile_2d(&WtS[(c + 1) & 1][0], Wstrip + (c + 1) * 256, 256,
                         64, 1024);
        __builtin_amdgcn_s_wait_tensorcnt((short)1);  // chunk c complete
      } else {
        __builtin_amdgcn_s_wait_tensorcnt((short)0);
      }
    }
    __syncthreads();                        // chunk c visible to all waves

    bf16x16 afr[8];
#pragma unroll
    for (int t = 0; t < 8; ++t)
      afr[t] = load_frag_a(X, 1024, m0, c * 256 + t * 32, lane);
#pragma unroll
    for (int t = 0; t < 8; ++t) {
#pragma unroll
      for (int j = 0; j < 4; ++j) {
        bf16x16 b =
            load_frag_b(&WtS[c & 1][0], 256, j * 16, (size_t)(t * 32), lane);
        acc[j] = wmma_bf16(afr[t], b, acc[j]);
      }
    }
    __syncthreads();                        // done reading before overwrite
  }

  const int h = lane >> 4, cl = lane & 15;
  if (sel < 2) {
    unsigned short* C = (sel == 0) ? Q : K;
#pragma unroll
    for (int j = 0; j < 4; ++j)
#pragma unroll
      for (int i = 0; i < 8; ++i)
        C[(size_t)(m0 + i + 8 * h) * 1024 + (n0 + j * 16 + cl)] =
            f2bf(acc[j][i]);
  } else {
#pragma unroll
    for (int j = 0; j < 4; ++j)
#pragma unroll
      for (int i = 0; i < 8; ++i)
        Vt[(size_t)(n0 + j * 16 + cl) * 8192 + (m0 + i + 8 * h)] =
            f2bf(acc[j][i]);
  }
}

// ---------------------------------------------------------------------------
// Stage 2: fused attention. One 32-thread block (1 wave) per 16 query rows.
// LDS: 16x2048 bf16 score strip (64KB) + 2 x 64x128 bf16 TDM tiles (32KB).
// K tiles (phase 1) and Vt tiles (phase 3) stream through the TDM pipeline:
// issue(t+1) -> s_wait_tensorcnt<=1 -> compute(t). Single-wave WG: no
// barriers needed inside the pipeline.
// ---------------------------------------------------------------------------
__global__ __launch_bounds__(32) void sattn_flash(
    const unsigned short* __restrict__ Q,
    const unsigned short* __restrict__ K,
    const unsigned short* __restrict__ Vt,
    float* __restrict__ out) {
  __shared__ __align__(16) unsigned short P[16 * 2048];      // 64KB
  __shared__ __align__(16) unsigned short TB[2][64 * 128];   // 2 x 16KB

  const int lane = threadIdx.x;
  const int b  = blockIdx.y;
  const int q0 = blockIdx.x * 16;
  const unsigned short* Qb = Q + (size_t)b * 2048 * 1024;
  const unsigned short* Kb = K + (size_t)b * 2048 * 1024;
  const int h = lane >> 4, cl = lane & 15;

  // ---- Phase 1: S = (Q K^T) * 1/sqrt(1024) -> LDS bf16 ------------------
  // Tiles t = nt*8+kc : K[nt*64 .. +63][kc*128 .. +127]
  tdm_load_tile_2d(&TB[0][0], Kb, 128, 64, 1024);
  for (int nt = 0; nt < 32; ++nt) {
    f32x8 acc[4] = {};
    for (int kc = 0; kc < 8; ++kc) {
      const int t = nt * 8 + kc;
      if (t + 1 < 256) {
        const int nt2 = (t + 1) >> 3, kc2 = (t + 1) & 7;
        tdm_load_tile_2d(&TB[(t + 1) & 1][0],
                         Kb + (size_t)(nt2 * 64) * 1024 + kc2 * 128, 128, 64,
                         1024);
        __builtin_amdgcn_s_wait_tensorcnt((short)1);
      } else {
        __builtin_amdgcn_s_wait_tensorcnt((short)0);
      }
      const unsigned short* kb = &TB[t & 1][0];
      bf16x16 afr[4];
#pragma unroll
      for (int u = 0; u < 4; ++u)
        afr[u] = load_frag_a(Qb, 1024, q0, kc * 128 + u * 32, lane);
#pragma unroll
      for (int u = 0; u < 4; ++u)
#pragma unroll
        for (int j = 0; j < 4; ++j) {
          bf16x16 bb = load_frag_b(kb, 128, j * 16, (size_t)(u * 32), lane);
          acc[j] = wmma_bf16(afr[u], bb, acc[j]);
        }
    }
#pragma unroll
    for (int j = 0; j < 4; ++j)
#pragma unroll
      for (int i = 0; i < 8; ++i)
        P[(i + 8 * h) * 2048 + nt * 64 + j * 16 + cl] =
            f2bf(acc[j][i] * 0.03125f);
  }
  __syncthreads();

  // ---- Phase 2: row softmax in LDS (2 lanes/row, shfl_xor combine) -------
  float rinv;  // 1/rowsum for row (lane & 15)
  {
    unsigned short* prow = &P[(lane & 15) * 2048 + (lane >> 4) * 1024];
    float mx = -3.0e38f;
    for (int j0 = 0; j0 < 1024; j0 += 8) {
      uint4 qv = *(const uint4*)&prow[j0];
      const unsigned* w = (const unsigned*)&qv;
#pragma unroll
      for (int t = 0; t < 4; ++t) {
        mx = fmaxf(mx, bf2f((unsigned short)(w[t] & 0xffffu)));
        mx = fmaxf(mx, bf2f((unsigned short)(w[t] >> 16)));
      }
    }
    mx = fmaxf(mx, __shfl_xor(mx, 16, 32));
    float sum = 0.f;
    for (int j0 = 0; j0 < 1024; j0 += 8) {
      uint4 qv = *(const uint4*)&prow[j0];
      unsigned* w = (unsigned*)&qv;
#pragma unroll
      for (int t = 0; t < 4; ++t) {
        float e0 = __expf(bf2f((unsigned short)(w[t] & 0xffffu)) - mx);
        float e1 = __expf(bf2f((unsigned short)(w[t] >> 16)) - mx);
        sum += e0 + e1;
        w[t] = (unsigned)f2bf(e0) | ((unsigned)f2bf(e1) << 16);
      }
      *(uint4*)&prow[j0] = qv;
    }
    sum += __shfl_xor(sum, 16, 32);
    rinv = 1.0f / sum;
  }
  __syncthreads();

  // ---- Phase 3: O = P * V.  Tiles t = c*16+ks : Vt[c*64..+63][ks*128..] --
  tdm_load_tile_2d(&TB[0][0], Vt + (size_t)b * 2048, 128, 64, 8192);
  for (int c = 0; c < 16; ++c) {
    f32x8 acc[4] = {};
    for (int ks = 0; ks < 16; ++ks) {
      const int t = c * 16 + ks;
      if (t + 1 < 256) {
        const int c2 = (t + 1) >> 4, ks2 = (t + 1) & 15;
        tdm_load_tile_2d(&TB[(t + 1) & 1][0],
                         Vt + (size_t)(c2 * 64) * 8192 + b * 2048 + ks2 * 128,
                         128, 64, 8192);
        __builtin_amdgcn_s_wait_tensorcnt((short)1);
      } else {
        __builtin_amdgcn_s_wait_tensorcnt((short)0);
      }
      const unsigned short* vb = &TB[t & 1][0];
      bf16x16 afr[4];
#pragma unroll
      for (int u = 0; u < 4; ++u)
        afr[u] = load_frag_a(P, 2048, 0, ks * 128 + u * 32, lane);
#pragma unroll
      for (int u = 0; u < 4; ++u)
#pragma unroll
        for (int j = 0; j < 4; ++j) {
          bf16x16 bb = load_frag_b(vb, 128, j * 16, (size_t)(u * 32), lane);
          acc[j] = wmma_bf16(afr[u], bb, acc[j]);
        }
    }
#pragma unroll
    for (int j = 0; j < 4; ++j) {
      const int col = c * 64 + j * 16 + cl;
#pragma unroll
      for (int i = 0; i < 8; ++i) {
        const int r = i + 8 * h;
        const float rs = __shfl(rinv, r, 32);  // lane r holds row r's 1/sum
        out[(size_t)(b * 2048 + q0 + r) * 1024 + col] = acc[j][i] * rs;
      }
    }
  }
}

// ---------------------------------------------------------------------------
// Launch. Workspace (bytes): [0,16M) Xbf | [16M,22M) Wt q/k/v | [22M,38M) Q |
// [38M,54M) K | [54M,70M) Vt   (all bf16)
// ---------------------------------------------------------------------------
extern "C" void kernel_launch(void* const* d_in, const int* in_sizes, int n_in,
                              void* d_out, int out_size, void* d_ws,
                              size_t ws_size, hipStream_t stream) {
  (void)in_sizes; (void)n_in; (void)out_size; (void)ws_size;
  const float* x  = (const float*)d_in[0];
  const float* Wq = (const float*)d_in[1];
  const float* Wk = (const float*)d_in[2];
  const float* Wv = (const float*)d_in[3];
  float* out = (float*)d_out;

  char* ws = (char*)d_ws;
  unsigned short* Xb  = (unsigned short*)(ws);
  unsigned short* Wtq = (unsigned short*)(ws + (16u << 20));
  unsigned short* Wtk = (unsigned short*)(ws + (18u << 20));
  unsigned short* Wtv = (unsigned short*)(ws + (20u << 20));
  unsigned short* Qm  = (unsigned short*)(ws + (22u << 20));
  unsigned short* Km  = (unsigned short*)(ws + (38u << 20));
  unsigned short* Vt  = (unsigned short*)(ws + (54u << 20));

  const int nx = 4 * 2048 * 1024;  // 8388608
  sattn_cast_x<<<dim3((nx + 255) / 256), dim3(256), 0, stream>>>(x, Xb, nx);
  sattn_cast_wt<<<dim3(4096, 1, 3), dim3(256), 0, stream>>>(Wq, Wk, Wv, Wtq,
                                                            Wtk, Wtv);
  // 64 m-groups (8 tiles each) x 16 n-strips, z = 3 matrices
  sattn_qkv_gemm<<<dim3(1024, 1, 3), dim3(256), 0, stream>>>(Xb, Wtq, Wtk,
                                                             Wtv, Qm, Km, Vt);
  // 128 query tiles x 4 batches, one wave each
  sattn_flash<<<dim3(128, 4), dim3(32), 0, stream>>>(Qm, Km, Vt, out);
}
